// Model_46926812676814
// MI455X (gfx1250) — matrix operand order repaired and checked by reading, not compiled, not run on hardware
//
#include <hip/hip_runtime.h>
#include <math.h>

typedef __attribute__((ext_vector_type(16))) _Float16 v16h;
typedef __attribute__((ext_vector_type(8)))  float    v8f;

#define Bsz 64
#define Lsz 512
#define Dsz 128
#define NHn 8
#define ELn 2
#define PLn 96
#define CINn 8
#define DFFn 2048
#define NTFn 4

__device__ __forceinline__ v8f wmma_f16(v16h a, v16h b, v8f c) {
  // D = A(16x32 f16) * B(32x16 f16) + C(16x16 f32)
  return __builtin_amdgcn_wmma_f32_16x16x32_f16(false, a, false, b, (short)0, c, false, false);
}
__device__ __forceinline__ float sigf(float x) { return 1.0f / (1.0f + __expf(-x)); }

// ---------------------------------------------------------------------------
// GARCH scan: sequential per batch element (64 lanes total)
// ---------------------------------------------------------------------------
__global__ void garch_kernel(const float* __restrict__ xe,
                             const float* __restrict__ gom, const float* __restrict__ gal,
                             const float* __restrict__ gbe, const float* __restrict__ gh0,
                             float* __restrict__ out) {
  int b = blockIdx.x * blockDim.x + threadIdx.x;
  if (b >= Bsz) return;
  float omega = log1pf(expf(gom[0]));
  float alpha = 0.2f * sigf(gal[0]);
  float beta  = 0.8f * sigf(gbe[0]);
  float h = log1pf(expf(gh0[0]));
  for (int t = 0; t < Lsz; ++t) {
    float r = xe[((size_t)b * Lsz + t) * CINn + (CINn - 2)];
    h = omega + alpha * r * r + beta * h;
  }
  for (int p = 0; p < PLn; ++p) {       // hs[0] = h_last, then iterate
    out[b * PLn + p] = sqrtf(h);
    h = omega + (alpha + beta) * h;
  }
}

// ---------------------------------------------------------------------------
// Embedding: circular conv1d(k=3) + temporal proj + positional encoding
// ---------------------------------------------------------------------------
__global__ void embed_kernel(const float* __restrict__ xe, const float* __restrict__ xm,
                             const float* __restrict__ cw, const float* __restrict__ tw,
                             float* __restrict__ enc) {
  int idx = blockIdx.x * 256 + threadIdx.x;           // B*L*D
  if (idx >= Bsz * Lsz * Dsz) return;
  int d = idx & (Dsz - 1);
  int l = (idx >> 7) & (Lsz - 1);
  int b = idx >> 16;
  float s = 0.f;
  for (int j = 0; j < 3; ++j) {
    int lm = (l + j - 1 + Lsz) & (Lsz - 1);           // wrap pad
    const float* xr = &xe[((size_t)b * Lsz + lm) * CINn];
    for (int ci = 0; ci < CINn; ++ci) s += xr[ci] * cw[(d * CINn + ci) * 3 + j];
  }
  const float* mr = &xm[((size_t)b * Lsz + l) * NTFn];
  for (int t = 0; t < NTFn; ++t) s += mr[t] * tw[d * NTFn + t];
  int i2 = d >> 1;
  float freq = __expf(-(float)(2 * i2) * (logf(10000.f) / (float)Dsz));
  float ang = (float)l * freq;
  s += (d & 1) ? cosf(ang) : sinf(ang);
  enc[idx] = s;
}

// ---------------------------------------------------------------------------
// Generic WMMA GEMM: C[M,N] = op( A[M,K] @ W[N,K]^T + bias1 + bias2 (+C) )
// Block tile 128x64, 8 waves as 4(row)x2(col); each wave owns a 32x32 output
// (4 accumulators) -> 4 WMMAs per wave per 32-wide k-step from 2 A-frags x
// 2 B-frags. LDS is double-buffered (one barrier per k-step; staging overlaps
// WMMA issue). M%128==0, N%64==0, K%32==0 guaranteed by callers.
// ---------------------------------------------------------------------------
__global__ void __launch_bounds__(256) gemm_wmma(
    const float* __restrict__ A, const float* __restrict__ W,
    const float* __restrict__ bias1, const float* __restrict__ bias2,
    float* __restrict__ C, int M, int N, int K,
    int lda, int ldw, int ldc, int relu, int accum) {
  __shared__ _Float16 As[2][128][32];   // 16 KB
  __shared__ _Float16 Bs[2][64][32];    //  8 KB
  int tid = threadIdx.x, lane = tid & 31, wv = tid >> 5;
  int nblk = N >> 6;
  int m0 = (blockIdx.x / nblk) << 7;
  int n0 = (blockIdx.x % nblk) << 6;
  int wr = wv & 3, wc = wv >> 2;        // 4 row-slabs x 2 col-slabs
  v8f acc00 = {}, acc01 = {}, acc10 = {}, acc11 = {};
  const int T = K >> 5;

  auto stage = [&](int kt, int buf) {
    int kk = kt << 5;
    for (int e = 0; e < 16; ++e) {                 // A: 128x32, 16/thread (64B runs)
      int idx = tid * 16 + e;
      int i = idx >> 5, j = idx & 31;
      As[buf][i][j] = (_Float16)A[(size_t)(m0 + i) * lda + kk + j];
    }
    for (int e = 0; e < 8; ++e) {                  // B: 64x32, 8/thread (32B runs)
      int idx = tid * 8 + e;
      int i = idx >> 5, j = idx & 31;
      Bs[buf][i][j] = (_Float16)W[(size_t)(n0 + i) * ldw + kk + j];
    }
    // speculative prefetch of the slab after next (dropped if OOB)
    __builtin_prefetch(&A[(size_t)(m0 + (tid >> 1)) * lda + kk + 32], 0, 0);
  };

  stage(0, 0);
  __syncthreads();
  for (int kt = 0; kt < T; ++kt) {
    int cur = kt & 1;
    if (kt + 1 < T) stage(kt + 1, cur ^ 1);
    int r = lane & 15;
    v16h a0, a1, b0, b1;
    {
      int koff = (lane < 16) ? 0 : 16;
      for (int h = 0; h < 16; ++h) {
        int kx = koff + h;
        a0[h] = As[cur][wr * 32 + r][kx];
        a1[h] = As[cur][wr * 32 + 16 + r][kx];
      }
      int kb = (lane < 16) ? 0 : 16;
      for (int h = 0; h < 16; ++h) {
        b0[h] = Bs[cur][wc * 32 + r][kb + h];
        b1[h] = Bs[cur][wc * 32 + 16 + r][kb + h];
      }
    }
    acc00 = wmma_f16(a0, b0, acc00);
    acc01 = wmma_f16(a0, b1, acc01);
    acc10 = wmma_f16(a1, b0, acc10);
    acc11 = wmma_f16(a1, b1, acc11);
    __syncthreads();
  }

  int nA = n0 + wc * 32 + (lane & 15);
  int nB = nA + 16;
  float bvA = 0.f, bvB = 0.f;
  if (bias1) { bvA += bias1[nA]; bvB += bias1[nB]; }
  if (bias2) { bvA += bias2[nA]; bvB += bias2[nB]; }
  int mb = m0 + wr * 32 + ((lane < 16) ? 0 : 8);
  for (int r2 = 0; r2 < 8; ++r2) {
    float v00 = acc00[r2] + bvA, v01 = acc01[r2] + bvB;
    float v10 = acc10[r2] + bvA, v11 = acc11[r2] + bvB;
    size_t i00 = (size_t)(mb + r2) * ldc + nA;
    size_t i01 = (size_t)(mb + r2) * ldc + nB;
    size_t i10 = (size_t)(mb + 16 + r2) * ldc + nA;
    size_t i11 = (size_t)(mb + 16 + r2) * ldc + nB;
    if (accum) { v00 += C[i00]; v01 += C[i01]; v10 += C[i10]; v11 += C[i11]; }
    if (relu) {
      v00 = fmaxf(v00, 0.f); v01 = fmaxf(v01, 0.f);
      v10 = fmaxf(v10, 0.f); v11 = fmaxf(v11, 0.f);
    }
    C[i00] = v00; C[i01] = v01; C[i10] = v10; C[i11] = v11;
  }
}

// ---------------------------------------------------------------------------
// LSTM recurrence. 4 blocks x 16 batch rows. Whh kept as WMMA B-fragments in
// VGPRs (16 v16h per wave); h state in LDS as f16 (A-fragment source).
// Per step: g = xW[:,t,:] + h @ Whh^T  (16x512 = 32 tiles, 4/wave, K=128)
// ---------------------------------------------------------------------------
__global__ void __launch_bounds__(256) lstm_kernel(
    const float* __restrict__ xW,    // [B*L][4D]
    const float* __restrict__ Whh,   // [4D][D]
    float* __restrict__ xout) {      // [B*L][D]
  __shared__ _Float16 hbuf[16][Dsz];
  __shared__ float    cbuf[16][Dsz];
  __shared__ float    gbuf[16][4 * Dsz];
  int tid = threadIdx.x, lane = tid & 31, wv = tid >> 5;
  int b0 = blockIdx.x * 16;
  for (int e = tid; e < 16 * Dsz; e += 256) {
    hbuf[e >> 7][e & 127] = (_Float16)0.f;
    cbuf[e >> 7][e & 127] = 0.f;
  }
  // hoist Whh B-fragments: wave wv owns output cols [wv*64, wv*64+64)
  v16h bfr[4][4];
  for (int c = 0; c < 4; ++c) {
    int n = wv * 64 + c * 16 + (lane & 15);
    for (int kc = 0; kc < 4; ++kc) {
      int kb = kc * 32 + ((lane < 16) ? 0 : 16);
      v16h tb;
      for (int h = 0; h < 16; ++h) tb[h] = (_Float16)Whh[(size_t)n * Dsz + kb + h];
      bfr[c][kc] = tb;
    }
  }
  __syncthreads();
  for (int t = 0; t < Lsz; ++t) {
    v16h af[4];
    {
      int r = lane & 15;
      for (int kc = 0; kc < 4; ++kc)
        for (int h = 0; h < 16; ++h) {
          int kx = kc * 32 + ((lane < 16) ? 0 : 16) + h;
          af[kc][h] = hbuf[r][kx];
        }
    }
    for (int c = 0; c < 4; ++c) {
      v8f acc = {};
      for (int kc = 0; kc < 4; ++kc) acc = wmma_f16(af[kc], bfr[c][kc], acc);
      int n = wv * 64 + c * 16 + (lane & 15);
      int mb = (lane < 16) ? 0 : 8;
      for (int r = 0; r < 8; ++r) {
        int m = mb + r;
        gbuf[m][n] = acc[r] + xW[((size_t)(b0 + m) * Lsz + t) * (4 * Dsz) + n];
      }
    }
    __syncthreads();
    for (int e = tid; e < 16 * Dsz; e += 256) {
      int m = e >> 7, d = e & 127;
      float gi = gbuf[m][d], gf = gbuf[m][Dsz + d];
      float gg = gbuf[m][2 * Dsz + d], go = gbuf[m][3 * Dsz + d];
      float cc = sigf(gf) * cbuf[m][d] + sigf(gi) * tanhf(gg);
      cbuf[m][d] = cc;
      float hv = sigf(go) * tanhf(cc);
      hbuf[m][d] = (_Float16)hv;
      xout[((size_t)(b0 + m) * Lsz + t) * Dsz + d] = hv;
    }
    __syncthreads();
  }
}

// ---------------------------------------------------------------------------
// Flash attention per (b, head): dh=16 (K padded to 32 with zeros).
// 8 waves, each owns 4 query tiles of 16 rows; K/V staged in LDS as f16;
// online softmax with cross-lane shuffles; P->A-frag roundtrip via LDS.
// ---------------------------------------------------------------------------
__global__ void __launch_bounds__(256) attn_kernel(
    const float* __restrict__ qkv,   // [B*L][3D]
    float* __restrict__ ctxo) {      // [B*L][D]
  __shared__ _Float16 Ks[Lsz][16];
  __shared__ _Float16 Vs[Lsz][16];
  __shared__ _Float16 pbuf[8][16][32];
  int tid = threadIdx.x, lane = tid & 31, wv = tid >> 5;
  int b = blockIdx.x >> 3;
  int hh = blockIdx.x & 7;
  size_t base = (size_t)b * Lsz;
  for (int e = tid; e < Lsz * 16; e += 256) {
    int key = e >> 4, d = e & 15;
    Ks[key][d] = (_Float16)qkv[(base + key) * (3 * Dsz) + Dsz + hh * 16 + d];
    Vs[key][d] = (_Float16)qkv[(base + key) * (3 * Dsz) + 2 * Dsz + hh * 16 + d];
  }
  for (int e = lane; e < 16 * 16; e += 32)   // upper K-half of P stays zero
    pbuf[wv][e >> 4][16 + (e & 15)] = (_Float16)0.f;
  __syncthreads();
  for (int qt = 0; qt < 4; ++qt) {
    int q0 = (wv * 4 + qt) * 16;
    v16h qa;
    {
      int r = lane & 15;
      for (int h = 0; h < 16; ++h) {
        int kx = ((lane < 16) ? 0 : 16) + h;
        qa[h] = (kx < 16) ? (_Float16)qkv[(base + q0 + r) * (3 * Dsz) + hh * 16 + kx]
                          : (_Float16)0.f;
      }
    }
    float mrow[8], lrow[8];
    for (int r = 0; r < 8; ++r) { mrow[r] = -1e30f; lrow[r] = 0.f; }
    v8f acc = {};
    for (int kj = 0; kj < Lsz / 16; ++kj) {
      int k0 = kj * 16;
      v16h kb;
      {
        int kcol = k0 + (lane & 15);
        for (int h = 0; h < 16; ++h) {
          int kx = h + ((lane < 16) ? 0 : 16);
          kb[h] = (kx < 16) ? Ks[kcol][kx] : (_Float16)0.f;
        }
      }
      v8f s = {};
      s = wmma_f16(qa, kb, s);
      for (int r = 0; r < 8; ++r) {
        float sv = s[r] * 0.25f;                   // 1/sqrt(dh)
        float mx = sv;
        mx = fmaxf(mx, __shfl_xor(mx, 1));
        mx = fmaxf(mx, __shfl_xor(mx, 2));
        mx = fmaxf(mx, __shfl_xor(mx, 4));
        mx = fmaxf(mx, __shfl_xor(mx, 8));
        float mnew = fmaxf(mrow[r], mx);
        float sc = __expf(mrow[r] - mnew);
        float p  = __expf(sv - mnew);
        float ps = p;
        ps += __shfl_xor(ps, 1);
        ps += __shfl_xor(ps, 2);
        ps += __shfl_xor(ps, 4);
        ps += __shfl_xor(ps, 8);
        lrow[r] = lrow[r] * sc + ps;
        mrow[r] = mnew;
        acc[r] *= sc;
        pbuf[wv][r + ((lane < 16) ? 0 : 8)][lane & 15] = (_Float16)p;
      }
      __syncthreads();
      v16h pa;
      {
        int r = lane & 15;
        for (int h = 0; h < 16; ++h) {
          int kx = ((lane < 16) ? 0 : 16) + h;
          pa[h] = pbuf[wv][r][kx];
        }
      }
      v16h vb;
      {
        int n = lane & 15;
        for (int h = 0; h < 16; ++h) {
          int kx = h + ((lane < 16) ? 0 : 16);
          vb[h] = (kx < 16) ? Vs[k0 + kx][n] : (_Float16)0.f;
        }
      }
      acc = wmma_f16(pa, vb, acc);
      __syncthreads();
    }
    int n = lane & 15;
    int mb = (lane < 16) ? 0 : 8;
    for (int r = 0; r < 8; ++r)
      ctxo[(base + q0 + mb + r) * Dsz + hh * 16 + n] = acc[r] / lrow[r];
  }
}

// ---------------------------------------------------------------------------
// x = LayerNorm(x + t) * g + b   (one 128-thread block per row)
// ---------------------------------------------------------------------------
__global__ void __launch_bounds__(128) addln_kernel(
    float* __restrict__ x, const float* __restrict__ t,
    const float* __restrict__ g, const float* __restrict__ bta) {
  __shared__ float red[Dsz];
  int row = blockIdx.x, d = threadIdx.x;
  size_t i = (size_t)row * Dsz + d;
  float v = x[i] + t[i];
  red[d] = v; __syncthreads();
  for (int s = 64; s > 0; s >>= 1) { if (d < s) red[d] += red[d + s]; __syncthreads(); }
  float mu = red[0] / (float)Dsz;
  __syncthreads();
  float dv = v - mu;
  red[d] = dv * dv; __syncthreads();
  for (int s = 64; s > 0; s >>= 1) { if (d < s) red[d] += red[d + s]; __syncthreads(); }
  float var = red[0] / (float)Dsz;
  x[i] = dv * rsqrtf(var + 1e-5f) * g[d] + bta[d];
}

// ---------------------------------------------------------------------------
// Head: ai = last@head_w^T+b ; gate = sig(relu(last@gw1^T+b)@gw2^T+b)
// final = garch + gate*ai     (tiny: one block per batch element)
// ---------------------------------------------------------------------------
__global__ void __launch_bounds__(128) head_kernel(
    const float* __restrict__ x, const float* __restrict__ hw, const float* __restrict__ hb,
    const float* __restrict__ gw1, const float* __restrict__ gb1,
    const float* __restrict__ gw2, const float* __restrict__ gb2,
    const float* __restrict__ garch, float* __restrict__ out) {
  __shared__ float last[Dsz];
  __shared__ float gh[Dsz / 2];
  int b = blockIdx.x, t = threadIdx.x;
  last[t] = x[((size_t)b * Lsz + (Lsz - 1)) * Dsz + t];
  __syncthreads();
  if (t < Dsz / 2) {
    float s = gb1[t];
    for (int k = 0; k < Dsz; ++k) s += gw1[t * Dsz + k] * last[k];
    gh[t] = fmaxf(s, 0.f);
  }
  __syncthreads();
  if (t < PLn) {
    float ai = hb[t];
    for (int k = 0; k < Dsz; ++k) ai += hw[t * Dsz + k] * last[k];
    float gv = gb2[t];
    for (int k = 0; k < Dsz / 2; ++k) gv += gw2[t * (Dsz / 2) + k] * gh[k];
    out[b * PLn + t] = garch[b * PLn + t] + sigf(gv) * ai;
  }
}

// ---------------------------------------------------------------------------
extern "C" void kernel_launch(void* const* d_in, const int* in_sizes, int n_in,
                              void* d_out, int out_size, void* d_ws, size_t ws_size,
                              hipStream_t stream) {
  (void)in_sizes; (void)n_in; (void)out_size; (void)ws_size;
  const float* x_enc      = (const float*)d_in[0];
  const float* x_mark_enc = (const float*)d_in[1];
  const float* conv_w     = (const float*)d_in[4];
  const float* temp_w     = (const float*)d_in[5];
  const float* lstm_Wih   = (const float*)d_in[6];
  const float* lstm_Whh   = (const float*)d_in[7];
  const float* lstm_bih   = (const float*)d_in[8];
  const float* lstm_bhh   = (const float*)d_in[9];
  const float* Wqkv       = (const float*)d_in[10];
  const float* bqkv       = (const float*)d_in[11];
  const float* Wo         = (const float*)d_in[12];
  const float* bo         = (const float*)d_in[13];
  const float* ln1g       = (const float*)d_in[14];
  const float* ln1b       = (const float*)d_in[15];
  const float* W1         = (const float*)d_in[16];
  const float* b1         = (const float*)d_in[17];
  const float* W2         = (const float*)d_in[18];
  const float* b2         = (const float*)d_in[19];
  const float* ln2g       = (const float*)d_in[20];
  const float* ln2b       = (const float*)d_in[21];
  const float* head_w     = (const float*)d_in[22];
  const float* head_b     = (const float*)d_in[23];
  const float* gate_w1    = (const float*)d_in[24];
  const float* gate_b1    = (const float*)d_in[25];
  const float* gate_w2    = (const float*)d_in[26];
  const float* gate_b2    = (const float*)d_in[27];
  const float* g_omega    = (const float*)d_in[28];
  const float* g_alpha    = (const float*)d_in[29];
  const float* g_beta     = (const float*)d_in[30];
  const float* g_h0       = (const float*)d_in[31];
  float* out = (float*)d_out;

  // Workspace layout (bytes): [0,16M) x/enc ; [16M,80M) big (xW, later ctx|tmp|ffbuf) ;
  // [80M,128M) qkv ; [128M,+24K) garch.
  char* ws = (char*)d_ws;
  float* buf_x   = (float*)ws;                       // B*L*D
  float* buf_big = (float*)(ws + (size_t)(16u << 20));
  float* buf_qkv = (float*)(ws + (size_t)(80u << 20));
  float* buf_gar = (float*)(ws + (size_t)(128u << 20));
  float* ctx  = buf_big;                             // B*L*D    (16 MB)
  float* tmp  = buf_big + ((size_t)4u << 20);        // B*L*D    (16 MB)
  float* ffb  = buf_big + ((size_t)8u << 20);        // B*L*256  (32 MB)

  const int M = Bsz * Lsz;                           // 32768
  const int MB = M / 128;                            // 256 row-blocks

  garch_kernel<<<1, 64, 0, stream>>>(x_enc, g_omega, g_alpha, g_beta, g_h0, buf_gar);
  embed_kernel<<<(Bsz * Lsz * Dsz) / 256, 256, 0, stream>>>(x_enc, x_mark_enc, conv_w, temp_w, buf_x);

  // xW = enc @ Wih^T + (bih + bhh)  -> buf_big [M x 512]
  gemm_wmma<<<MB * (4 * Dsz / 64), 256, 0, stream>>>(
      buf_x, lstm_Wih, lstm_bih, lstm_bhh, buf_big,
      M, 4 * Dsz, Dsz, Dsz, Dsz, 4 * Dsz, 0, 0);

  // LSTM recurrence -> x (hseq) in buf_x
  lstm_kernel<<<Bsz / 16, 256, 0, stream>>>(buf_big, lstm_Whh, buf_x);

  for (int li = 0; li < ELn; ++li) {
    // QKV projection
    gemm_wmma<<<MB * (3 * Dsz / 64), 256, 0, stream>>>(
        buf_x, Wqkv + (size_t)li * 3 * Dsz * Dsz, bqkv + (size_t)li * 3 * Dsz, nullptr,
        buf_qkv, M, 3 * Dsz, Dsz, Dsz, Dsz, 3 * Dsz, 0, 0);
    // attention
    attn_kernel<<<Bsz * NHn, 256, 0, stream>>>(buf_qkv, ctx);
    // output projection
    gemm_wmma<<<MB * (Dsz / 64), 256, 0, stream>>>(
        ctx, Wo + (size_t)li * Dsz * Dsz, bo + (size_t)li * Dsz, nullptr,
        tmp, M, Dsz, Dsz, Dsz, Dsz, Dsz, 0, 0);
    addln_kernel<<<M, 128, 0, stream>>>(buf_x, tmp, ln1g + li * Dsz, ln1b + li * Dsz);
    // FFN, chunked over DFF so the 256MB intermediate never exists
    for (int cf = 0; cf < DFFn / 256; ++cf) {
      gemm_wmma<<<MB * (256 / 64), 256, 0, stream>>>(
          buf_x, W1 + (size_t)li * DFFn * Dsz + (size_t)cf * 256 * Dsz,
          b1 + (size_t)li * DFFn + cf * 256, nullptr,
          ffb, M, 256, Dsz, Dsz, Dsz, 256, /*relu=*/1, 0);
      gemm_wmma<<<MB * (Dsz / 64), 256, 0, stream>>>(
          ffb, W2 + (size_t)li * Dsz * DFFn + (size_t)cf * 256,
          (cf == 0) ? (b2 + (size_t)li * Dsz) : nullptr, nullptr,
          tmp, M, Dsz, 256, 256, DFFn, Dsz, 0, /*accum=*/(cf > 0));
    }
    addln_kernel<<<M, 128, 0, stream>>>(buf_x, tmp, ln2g + li * Dsz, ln2b + li * Dsz);
  }

  head_kernel<<<Bsz, 128, 0, stream>>>(buf_x, head_w, head_b, gate_w1, gate_b1,
                                       gate_w2, gate_b2, buf_gar, out);
}
